// invariant_classifier_86535001080004
// MI455X (gfx1250) — matrix operand order repaired and checked by reading, not compiled, not run on hardware
//
#include <hip/hip_runtime.h>
#include <math.h>

#define Bn 2048
#define Hn 4096
#define En 8
#define Ln 512

typedef __attribute__((ext_vector_type(8)))  float          v8f;
typedef __attribute__((ext_vector_type(16))) __bf16         v16bf;
typedef __attribute__((ext_vector_type(4)))  unsigned short u16x4;
typedef __attribute__((ext_vector_type(8)))  unsigned short u16x8;
typedef __attribute__((ext_vector_type(16))) unsigned short u16x16;

// ---------- bf16 split helpers (native converts) ----------
__device__ __forceinline__ unsigned short f2bf(float f) {
    __bf16 h = (__bf16)f;                      // native v_cvt (RNE)
    return __builtin_bit_cast(unsigned short, h);
}
__device__ __forceinline__ float bf2f(unsigned short h) {
    return __uint_as_float(((unsigned)h) << 16);   // exact widening, 1 shift
}

__device__ __forceinline__ v16bf ldfrag(const unsigned short* p, int off0, int off1) {
    u16x8 a = *(const u16x8*)(p + off0);
    u16x8 b = *(const u16x8*)(p + off1);
    u16x16 t;
#pragma unroll
    for (int i = 0; i < 8; ++i) { t[i] = a[i]; t[i + 8] = b[i]; }
    return __builtin_bit_cast(v16bf, t);
}

__device__ __forceinline__ v8f wmma_bf16(v16bf a, v16bf b, v8f c) {
    return __builtin_amdgcn_wmma_f32_16x16x32_bf16(false, a, false, b, (short)0, c, false, false);
}

// ---------- 1. gating: argmax(envs+gumbel), gate = (1+s)-s ----------
__global__ void gate_kernel(const float* __restrict__ envs, const float* __restrict__ gum,
                            int* __restrict__ e_star, float* __restrict__ gate) {
    int b = blockIdx.x * blockDim.x + threadIdx.x;
    if (b >= Bn) return;
    float4 e0 = *(const float4*)(envs + b * En);
    float4 e1 = *(const float4*)(envs + b * En + 4);
    float4 g0 = *(const float4*)(gum + b * En);
    float4 g1 = *(const float4*)(gum + b * En + 4);
    float v[En] = { e0.x + g0.x, e0.y + g0.y, e0.z + g0.z, e0.w + g0.w,
                    e1.x + g1.x, e1.y + g1.y, e1.z + g1.z, e1.w + g1.w };
    float m = -3.4e38f; int am = 0;
#pragma unroll
    for (int e = 0; e < En; ++e) if (v[e] > m) { m = v[e]; am = e; }
    float sum = 0.f;
#pragma unroll
    for (int e = 0; e < En; ++e) sum += expf(v[e] - m);
    float s = 1.0f / sum;                 // y_soft at the argmax (max term is exp(0)=1)
    float g = (1.0f + s) - s;             // straight-through numeric value
    e_star[b] = am;
    gate[b] = g;
}

// ---------- 2. deterministic compaction by expert ----------
__global__ void compact_kernel(const int* __restrict__ e_star,
                               int* __restrict__ cnt, int* __restrict__ rows) {
    if (threadIdx.x == 0 && blockIdx.x == 0) {
        int c[En];
#pragma unroll
        for (int e = 0; e < En; ++e) c[e] = 0;
        for (int b = 0; b < Bn; ++b) {
            int e = e_star[b];
            rows[e * Bn + c[e]] = b;
            c[e]++;
        }
#pragma unroll
        for (int e = 0; e < En; ++e) cnt[e] = c[e];
    }
}

// ---------- 3. gather-GEMM: out[b, n] = gate[b] * hidden[b,:] . W[e, n, :] ----------
// Tiles: BM=64, BN=64, BK=64 (two K=32 WMMA chunks per staging round).
__global__ __launch_bounds__(256)
void moe_gemm(const float* __restrict__ hidden, const float* __restrict__ W,
              const int* __restrict__ rows, const int* __restrict__ cnt,
              const float* __restrict__ gate, float* __restrict__ out) {
    const int e = blockIdx.z;
    const int c = cnt[e];
    const int m0 = blockIdx.y * 64;
    if (m0 >= c) return;
    const int n0 = blockIdx.x * 64;
    const int tid = threadIdx.x;

    __shared__ unsigned short Ah[64 * 72];   // [M][K] bf16 hi, K-stride 72 (pad)
    __shared__ unsigned short Al[64 * 72];   // [M][K] bf16 lo
    __shared__ unsigned short Bh[64 * 72];   // [N][K] bf16 hi, K-stride 72 (pad)
    __shared__ unsigned short Bl[64 * 72];   // [N][K] bf16 lo
    __shared__ int rowIdx[64];

    if (tid < 64) {
        int mg = m0 + tid;
        rowIdx[tid] = rows[e * Bn + ((mg < c) ? mg : 0)];
    }
    __syncthreads();

    const float* We = W + (size_t)e * Ln * Hn;

    // ---- hoisted staging addresses (4 float4 per matrix per thread per round) ----
    const int sRow = tid >> 4;          // 0..15
    const int sK4  = (tid & 15) * 4;    // K base (floats) within 64-chunk
    const float* pA[4];
    const float* pB[4];
#pragma unroll
    for (int p = 0; p < 4; ++p) {
        pA[p] = hidden + (size_t)rowIdx[sRow + 16 * p] * Hn + sK4;
        pB[p] = We + (size_t)(n0 + sRow + 16 * p) * Hn + sK4;
    }

    const int wid  = tid >> 5;
    const int lane = tid & 31;
    const int wm = wid & 3;        // 4-way split of 64 M rows
    const int wn = wid >> 2;       // 2-way split of 64 N cols (32 each)

    // A fragment addressing (16-bit A 16x32 layout): lane<16 -> K{0..7,16..23}, lane>=16 -> +8
    const int aRow  = wm * 16 + (lane & 15);
    const int aKsel = (lane & 16) ? 8 : 0;

    v8f acc0, acc1;
#pragma unroll
    for (int i = 0; i < 8; ++i) { acc0[i] = 0.f; acc1[i] = 0.f; }

    float4 aReg[4], bReg[4];
#pragma unroll
    for (int p = 0; p < 4; ++p) {       // prologue fetch (k0 = 0)
        aReg[p] = *(const float4*)(pA[p]);
        bReg[p] = *(const float4*)(pB[p]);
    }

    for (int k0 = 0; k0 < Hn; k0 += 64) {
        // ---- commit regs -> LDS (previous round's frag reads are fenced below) ----
#pragma unroll
        for (int p = 0; p < 4; ++p) {
            int r = sRow + 16 * p;
            float f[4] = { aReg[p].x, aReg[p].y, aReg[p].z, aReg[p].w };
            u16x4 hv, lv;
#pragma unroll
            for (int j = 0; j < 4; ++j) {
                unsigned short h = f2bf(f[j]);
                hv[j] = h;
                lv[j] = f2bf(f[j] - bf2f(h));
            }
            *(u16x4*)(&Ah[r * 72 + sK4]) = hv;
            *(u16x4*)(&Al[r * 72 + sK4]) = lv;
        }
#pragma unroll
        for (int p = 0; p < 4; ++p) {   // commit for B: [n][k] -> [n][k]
            int n = sRow + 16 * p;
            float f[4] = { bReg[p].x, bReg[p].y, bReg[p].z, bReg[p].w };
#pragma unroll
            for (int j = 0; j < 4; ++j) {
                unsigned short h = f2bf(f[j]);
                Bh[n * 72 + sK4 + j] = h;
                Bl[n * 72 + sK4 + j] = f2bf(f[j] - bf2f(h));
            }
        }
        __syncthreads();

        // ---- issue next round's global loads early (latency hides behind WMMA) ----
        if (k0 + 64 < Hn) {
#pragma unroll
            for (int p = 0; p < 4; ++p) {
                aReg[p] = *(const float4*)(pA[p] + k0 + 64);
                bReg[p] = *(const float4*)(pB[p] + k0 + 64);
            }
        }

        // ---- compute both K=32 chunks ----
#pragma unroll
        for (int kc = 0; kc < 2; ++kc) {
            const int aO  = aRow * 72 + kc * 32 + aKsel;
            v16bf ah = ldfrag(Ah, aO, aO + 16);
            v16bf al = ldfrag(Al, aO, aO + 16);
            const int bO0 = (wn * 32 + (lane & 15)) * 72 + kc * 32 + aKsel;
            const int bO1 = bO0 + 16 * 72;
            v16bf bh0 = ldfrag(Bh, bO0, bO0 + 16);
            v16bf bl0 = ldfrag(Bl, bO0, bO0 + 16);
            v16bf bh1 = ldfrag(Bh, bO1, bO1 + 16);
            v16bf bl1 = ldfrag(Bl, bO1, bO1 + 16);

            // split-bf16 f32 emulation: ah*bh + ah*bl + al*bh
            acc0 = wmma_bf16(ah, bh0, acc0);
            acc0 = wmma_bf16(ah, bl0, acc0);
            acc0 = wmma_bf16(al, bh0, acc0);
            acc1 = wmma_bf16(ah, bh1, acc1);
            acc1 = wmma_bf16(ah, bl1, acc1);
            acc1 = wmma_bf16(al, bh1, acc1);
        }
        __syncthreads();   // LDS safe to overwrite next round
    }

    // epilogue: C layout lane%16 = N, VGPR r -> M = r + 8*(lane>>4)
    const int nlane = lane & 15;
    const int mhalf = (lane >> 4) * 8;
#pragma unroll
    for (int r = 0; r < 8; ++r) {
        int mloc = wm * 16 + mhalf + r;
        int mg = m0 + mloc;
        if (mg < c) {
            int gb = rowIdx[mloc];
            float g = gate[gb];
            size_t base = (size_t)gb * Ln + n0 + wn * 32 + nlane;
            out[base]      = g * acc0[r];
            out[base + 16] = g * acc1[r];
        }
    }
}

// ---------- 4. variance penalty over W ----------
#define PPOS (Ln * Hn)          // 2097152 positions
#define PBLK 1024

__global__ __launch_bounds__(256)
void penalty_partial(const float* __restrict__ W, float* __restrict__ partials) {
    __shared__ float red[256];
    float dsq[En], l1a[En];
#pragma unroll
    for (int e = 0; e < En; ++e) { dsq[e] = 0.f; l1a[e] = 0.f; }

    const int tglob = blockIdx.x * 256 + threadIdx.x;
    const int nthr = PBLK * 256;
    const float4* W4 = (const float4*)W;
    const int P4 = PPOS / 4;
    for (int p = tglob; p < P4; p += nthr) {
        float4 w[En];
        float mx = 0.f, my = 0.f, mz = 0.f, mw = 0.f;
#pragma unroll
        for (int e = 0; e < En; ++e) {
            w[e] = W4[(size_t)e * P4 + p];
            mx += w[e].x; my += w[e].y; mz += w[e].z; mw += w[e].w;
        }
        mx *= 0.125f; my *= 0.125f; mz *= 0.125f; mw *= 0.125f;
#pragma unroll
        for (int e = 0; e < En; ++e) {
            float dx = w[e].x - mx, dy = w[e].y - my, dz = w[e].z - mz, dw = w[e].w - mw;
            dsq[e] += dx * dx + dy * dy + dz * dz + dw * dw;
            l1a[e] += fabsf(w[e].x) + fabsf(w[e].y) + fabsf(w[e].z) + fabsf(w[e].w);
        }
    }
    for (int q = 0; q < 16; ++q) {
        red[threadIdx.x] = (q < En) ? dsq[q] : l1a[q - En];
        __syncthreads();
        for (int s = 128; s > 0; s >>= 1) {
            if (threadIdx.x < s) red[threadIdx.x] += red[threadIdx.x + s];
            __syncthreads();
        }
        if (threadIdx.x == 0) partials[blockIdx.x * 16 + q] = red[0];
        __syncthreads();
    }
}

__global__ void penalty_final(const float* __restrict__ partials, float* __restrict__ out_scalar) {
    __shared__ double acc[16];
    int tid = threadIdx.x;
    if (tid < 16) {
        double s = 0.0;
        for (int b = 0; b < PBLK; ++b) s += (double)partials[b * 16 + tid];
        acc[tid] = s;
    }
    __syncthreads();
    if (tid == 0) {
        double loss = 0.0;
        for (int e = 0; e < En; ++e) {
            double d = acc[e], l = acc[8 + e];
            loss += d / (l * l);
        }
        out_scalar[0] = (float)(loss * 0.125);
    }
}

// ---------- launch ----------
extern "C" void kernel_launch(void* const* d_in, const int* in_sizes, int n_in,
                              void* d_out, int out_size, void* d_ws, size_t ws_size,
                              hipStream_t stream) {
    const float* hidden = (const float*)d_in[0];
    const float* envs   = (const float*)d_in[1];
    // d_in[2] is idx = arange(B): identity gather, unused
    const float* gum    = (const float*)d_in[3];
    const float* W      = (const float*)d_in[4];
    float* out = (float*)d_out;   // [B*L] logits, then loss scalar

    char* ws = (char*)d_ws;
    int*   e_star   = (int*)ws;                         //  8 KB
    float* gate     = (float*)(ws + 8192);              //  8 KB
    int*   cnt      = (int*)(ws + 16384);               //  pad to 256 B
    int*   rows     = (int*)(ws + 16640);               // 64 KB
    float* partials = (float*)(ws + 16640 + 65536);     // 64 KB

    gate_kernel<<<Bn / 256, 256, 0, stream>>>(envs, gum, e_star, gate);
    compact_kernel<<<1, 1, 0, stream>>>(e_star, cnt, rows);

    dim3 grid(Ln / 64, Bn / 64, En);   // (8, 32, 8); m-tiles early-exit past cnt[e]
    moe_gemm<<<grid, 256, 0, stream>>>(hidden, W, rows, cnt, gate, out);

    penalty_partial<<<PBLK, 256, 0, stream>>>(W, partials);
    penalty_final<<<1, 256, 0, stream>>>(partials, out + (size_t)Bn * Ln);
}
